// SpectralLSTMCell_36223754174782
// MI455X (gfx1250) — matrix operand order repaired and checked by reading, not compiled, not run on hardware
//
#include <hip/hip_runtime.h>
#include <hip/hip_bf16.h>

#define H 512
#define I 64
#define B 64

typedef __attribute__((ext_vector_type(16))) __bf16 v16bf;
typedef __attribute__((ext_vector_type(8)))  __bf16 v8bf;
typedef __attribute__((ext_vector_type(8)))  float  v8f;

union Frag {
    v16bf v;
    unsigned int u[8];
    uint4 q[2];
};

union BF8 {
    v8bf v;
    uint4 q;
};

__device__ __forceinline__ unsigned short f2bf(float f) {
    unsigned int u = __builtin_bit_cast(unsigned int, f);
    unsigned int r = u + 0x7FFFu + ((u >> 16) & 1u);   // round-to-nearest-even
    return (unsigned short)(r >> 16);
}

__device__ __forceinline__ float sigmoidf(float v) {
    return 1.0f / (1.0f + __expf(-v));
}

// -------- kernel 0: zero the output accumulator --------
__global__ void zero_out_kernel(float* __restrict__ out, int n) {
    int i = blockIdx.x * blockDim.x + threadIdx.x;
    if (i < n) out[i] = 0.0f;
}

// -------- kernel 1: fp32 -> bf16 weight pre-conversion --------
__global__ void cvt_bf16_kernel(const float* __restrict__ src,
                                unsigned short* __restrict__ dst, int n) {
    int i = blockIdx.x * blockDim.x + threadIdx.x;
    if (i < n) dst[i] = f2bf(src[i]);
}

// -------- kernel 2: vector gate o = sigmoid(x@Wio^T + hx@Who^T + bo) --------
__global__ void o_gate_kernel(const float* __restrict__ x,
                              const float* __restrict__ hx,
                              const float* __restrict__ Wio,
                              const float* __restrict__ Who,
                              const float* __restrict__ bo,
                              float* __restrict__ o) {
    int idx = blockIdx.x * blockDim.x + threadIdx.x;   // b*H + c
    if (idx >= B * H) return;
    int b = idx >> 9;
    int c = idx & (H - 1);
    float s = bo[c];
    const float* xr = x + b * I;
    const float* wr = Wio + c * I;
    #pragma unroll 8
    for (int d = 0; d < I; ++d) s += xr[d] * wr[d];
    const float* hr = hx + b * H;
    const float* vr = Who + c * H;
    #pragma unroll 8
    for (int d = 0; d < H; ++d) s += hr[d] * vr[d];
    o[idx] = sigmoidf(s);
}

// -------- kernel 3: fused matrix gates + cell update + reduction --------
// grid: B * 64 blocks (8x8 tiles of the 512x512 gate matrix), 256 threads.
#define NSTEPS 18   // 2 K-steps of 32 for x-phase (K=64) + 16 for hx-phase (K=512)

__global__ __launch_bounds__(256)
void spectral_lstm_kernel(const float* __restrict__ x,
                          const float* __restrict__ hx,
                          const float* __restrict__ cx,
                          const unsigned short* __restrict__ wbf,  // 12 bf16 weights
                          const float* __restrict__ bi,
                          const float* __restrict__ bf,
                          const float* __restrict__ bg,
                          const float* __restrict__ oGate,
                          float* __restrict__ out) {
    __shared__ __align__(16) unsigned short As[2][3][64][32];  // ping-pong A tiles
    __shared__ __align__(16) unsigned short Bs[2][3][64][32];  // ping-pong B tiles
    __shared__ float red[4][64];
    __shared__ __align__(16) unsigned short xSb[I];
    __shared__ __align__(16) unsigned short hSb[H];
    __shared__ float oS[64];

    const int blk  = blockIdx.x;
    const int b    = blk >> 6;
    const int tile = blk & 63;
    const int tm   = (tile >> 3) * 64;    // c-row base
    const int tn   = (tile & 7) * 64;     // h-col base

    const int tid    = threadIdx.x;
    const int lane   = tid & 31;
    const int wid    = tid >> 5;
    const int wave_m = wid >> 1;          // 0..3 : 16-row band
    const int wave_n = wid & 1;           // 0..1 : 32-col band
    const int m16    = lane & 15;
    const int hi     = lane >> 4;

    const size_t HI = (size_t)H * I, HH = (size_t)H * H;
    // ws layout: Ax0 Ax1 Ax2 Bx0 Bx1 Bx2 Ah0 Ah1 Ah2 Bh0 Bh1 Bh2
    const unsigned short* wA[2][3] = {
        {wbf,            wbf + HI,           wbf + 2 * HI},
        {wbf + 6 * HI,   wbf + 6 * HI + HH,  wbf + 6 * HI + 2 * HH}};
    const unsigned short* wB[2][3] = {
        {wbf + 3 * HI,          wbf + 4 * HI,               wbf + 5 * HI},
        {wbf + 6 * HI + 3 * HH, wbf + 6 * HI + 4 * HH,      wbf + 6 * HI + 5 * HH}};

    // per-batch scale vectors (bf16) + o slice
    for (int t = tid; t < H; t += 256) hSb[t] = f2bf(hx[b * H + t]);
    if (tid < I)  xSb[tid] = f2bf(x[b * I + tid]);
    if (tid < 64) oS[tid] = oGate[b * H + tm + tid];
    __syncthreads();

    const int srow = tid >> 2;            // staging row 0..63
    const int scol = (tid & 3) * 8;       // staging col group (8 bf16 = 16B)

    // Stage one K-step: A = bf16 weights * diag(v) via packed bf16 mul + ds_store,
    // B = pure DMA global->LDS via async load (ASYNCcnt).
    auto stage = [&](int step, int bufi) {
        const int phs = (step < 2) ? 0 : 1;
        const int k0  = ((step < 2) ? step : step - 2) << 5;
        const int D   = phs ? H : I;
        const unsigned short* scb = phs ? hSb : xSb;
        BF8 s;
        s.q = *(const uint4*)&scb[k0 + scol];
        #pragma unroll
        for (int g = 0; g < 3; ++g) {
            BF8 w;
            w.q = *(const uint4*)(wA[phs][g] + (size_t)(tm + srow) * D + k0 + scol);
            w.v = w.v * s.v;                               // v_pk_mul_bf16
            *(uint4*)&As[bufi][g][srow][scol] = w.q;       // ds_store_b128
            const unsigned short* gb =
                wB[phs][g] + (size_t)(tn + srow) * D + k0 + scol;
            unsigned int ldst = (unsigned int)(size_t)&Bs[bufi][g][srow][scol];
            asm volatile("global_load_async_to_lds_b128 %0, %1, off"
                         :: "v"(ldst), "v"(gb) : "memory");
        }
    };

    v8f acc[3][2] = {};
    stage(0, 0);

    const int arow = wave_m * 16 + m16;
    const int brow = wave_n * 32 + m16;

    for (int step = 0; step < NSTEPS; ++step) {
        const int cur = step & 1;
        if (step + 1 < NSTEPS) {
            stage(step + 1, cur ^ 1);                      // prefetch next buffer
            asm volatile("s_wait_asynccnt 0x3" ::: "memory");  // oldest 3 (cur) done
        } else {
            asm volatile("s_wait_asynccnt 0x0" ::: "memory");
        }
        __syncthreads();
        // ---- WMMA on current buffer: 3 gates x 2 column subtiles ----
        #pragma unroll
        for (int g = 0; g < 3; ++g) {
            Frag af, b0, b1;
            af.q[0] = *(const uint4*)&As[cur][g][arow][hi * 16];
            af.q[1] = *(const uint4*)&As[cur][g][arow][hi * 16 + 8];
            const uint4* p0 = (const uint4*)&Bs[cur][g][brow][hi * 16];
            b0.q[0] = p0[0]; b0.q[1] = p0[1];
            const uint4* p1 = (const uint4*)&Bs[cur][g][brow + 16][hi * 16];
            b1.q[0] = p1[0]; b1.q[1] = p1[1];
            acc[g][0] = __builtin_amdgcn_wmma_f32_16x16x32_bf16(
                false, af.v, false, b0.v, (short)0, acc[g][0], false, false);
            acc[g][1] = __builtin_amdgcn_wmma_f32_16x16x32_bf16(
                false, af.v, false, b1.v, (short)0, acc[g][1], false, false);
        }
        __syncthreads();   // compute done before this buffer is restaged
    }

    // ---- epilogue: bias + sigmoid/tanh + cell update + o-weighted reduce ----
    const size_t cxB = (size_t)b * H * H;
    float colSum[2];
    #pragma unroll
    for (int nt = 0; nt < 2; ++nt) {
        const int nG = tn + wave_n * 32 + nt * 16 + m16;
        float s = 0.0f;
        #pragma unroll
        for (int r = 0; r < 8; ++r) {
            const int mL  = wave_m * 16 + hi * 8 + r;
            const int mG  = tm + mL;
            const size_t idx = (size_t)mG * H + nG;
            float iv = sigmoidf(acc[0][nt][r] + bi[idx]);
            float fv = sigmoidf(acc[1][nt][r] + bf[idx]);
            float gv = tanhf(acc[2][nt][r] + bg[idx]);
            float cn = fv * cx[cxB + idx] + iv * gv;
            s += tanhf(cn) * oS[mL];
        }
        s += __shfl_xor(s, 16, 32);                        // fold lane halves (rows)
        colSum[nt] = s;
    }
    if (hi == 0) {
        red[wave_m][wave_n * 32 + m16]      = colSum[0];
        red[wave_m][wave_n * 32 + 16 + m16] = colSum[1];
    }
    __syncthreads();
    if (tid < 64) {
        float s = red[0][tid] + red[1][tid] + red[2][tid] + red[3][tid];
        atomicAdd(out + b * H + tn + tid, s);              // global_atomic_add_f32
    }
}

extern "C" void kernel_launch(void* const* d_in, const int* in_sizes, int n_in,
                              void* d_out, int out_size, void* d_ws, size_t ws_size,
                              hipStream_t stream) {
    const float* x    = (const float*)d_in[0];
    const float* hx   = (const float*)d_in[1];
    const float* cx   = (const float*)d_in[2];
    const float* Wii  = (const float*)d_in[3];
    const float* Wiit = (const float*)d_in[4];
    const float* Whi  = (const float*)d_in[5];
    const float* Whit = (const float*)d_in[6];
    const float* bi   = (const float*)d_in[7];
    const float* Wif  = (const float*)d_in[8];
    const float* Wift = (const float*)d_in[9];
    const float* Whf  = (const float*)d_in[10];
    const float* Whft = (const float*)d_in[11];
    const float* bf   = (const float*)d_in[12];
    const float* Wig  = (const float*)d_in[13];
    const float* Wigt = (const float*)d_in[14];
    const float* Whc  = (const float*)d_in[15];
    const float* Whct = (const float*)d_in[16];
    const float* bg   = (const float*)d_in[17];
    const float* Wio  = (const float*)d_in[18];
    const float* Who  = (const float*)d_in[19];
    const float* bo   = (const float*)d_in[20];

    float* out = (float*)d_out;                         // [B,H]
    float* oW  = (float*)d_ws;                          // [B,H] vector-gate scratch
    unsigned short* wbf =
        (unsigned short*)((char*)d_ws + (size_t)B * H * sizeof(float));

    const size_t HI = (size_t)H * I, HH = (size_t)H * H;
    const float* srcs[12] = {Wii, Wif, Wig, Wiit, Wift, Wigt,
                             Whi, Whf, Whc, Whit, Whft, Whct};
    size_t off = 0;
    for (int i = 0; i < 12; ++i) {
        int n = (int)((i < 6) ? HI : HH);
        cvt_bf16_kernel<<<(n + 255) / 256, 256, 0, stream>>>(srcs[i], wbf + off, n);
        off += n;
    }

    zero_out_kernel<<<(B * H + 255) / 256, 256, 0, stream>>>(out, B * H);
    o_gate_kernel<<<(B * H + 255) / 256, 256, 0, stream>>>(x, hx, Wio, Who, bo, oW);
    spectral_lstm_kernel<<<B * 64, 256, 0, stream>>>(
        x, hx, cx, wbf, bi, bf, bg, oW, out);
}
